// PairAxisAttention_50044958933070
// MI455X (gfx1250) — compile-verified, matrix operand order checked by reading, not run on hardware
//
#include <hip/hip_runtime.h>
#include <hip/hip_bf16.h>

#define PAIR 128
#define NN 384
#define NHEADS 4
#define HS 32
#define NPOS (NN * NN) /* 147456 */

typedef __attribute__((ext_vector_type(16))) __bf16 bf16x16;
typedef __attribute__((ext_vector_type(8)))  __bf16 bf16x8;
typedef __attribute__((ext_vector_type(8)))  float  f32x8;
typedef __attribute__((ext_vector_type(4)))  float  f32x4;

// ---- WMMA helpers (CDNA5 16x16x32 bf16, f32 accum) -------------------------
// A (16x32, MxK) per-lane layout: row M = lane&15; elements e=0..15 map to
// K = (lane>=16 ? 8 : 0) + (e&7) + (e>=8 ? 16 : 0).  B (32x16, KxN) is loaded
// as N-major rows with the identical pattern (lane holds column n = lane&15).
__device__ __forceinline__ bf16x16 frag_ld(const __bf16* tile, int ld, int lane) {
  int r  = lane & 15;
  int kb = (lane >> 4) << 3;
  const __bf16* p = tile + r * ld + kb;
  bf16x8 lo = *(const bf16x8*)(p);
  bf16x8 hi = *(const bf16x8*)(p + 16);
  bf16x16 f;
#pragma unroll
  for (int e = 0; e < 8; ++e) { f[e] = lo[e]; f[e + 8] = hi[e]; }
  return f;
}

__device__ __forceinline__ bf16x16 frag_ld_f32(const float* tile, int ld, int lane) {
  int r  = lane & 15;
  int kb = (lane >> 4) << 3;
  const float* p = tile + r * ld + kb;
  f32x4 a0 = *(const f32x4*)(p);
  f32x4 a1 = *(const f32x4*)(p + 4);
  f32x4 b0 = *(const f32x4*)(p + 16);
  f32x4 b1 = *(const f32x4*)(p + 20);
  bf16x16 f;
#pragma unroll
  for (int e = 0; e < 4; ++e) {
    f[e]      = (__bf16)a0[e];
    f[e + 4]  = (__bf16)a1[e];
    f[e + 8]  = (__bf16)b0[e];
    f[e + 12] = (__bf16)b1[e];
  }
  return f;
}

__device__ __forceinline__ f32x8 wmma_bf16(bf16x16 a, bf16x16 b, f32x8 c) {
  return __builtin_amdgcn_wmma_f32_16x16x32_bf16(false, a, false, b, (short)0, c,
                                                 false, false);
}

// ---- 1) weight prep: W[in][out] f32 -> W^T[out][in] bf16 -------------------
__global__ __launch_bounds__(256) void prep_kernel(
    const float* __restrict__ Wq, const float* __restrict__ Wk,
    const float* __restrict__ Wv, const float* __restrict__ Wg,
    const float* __restrict__ Wo, __bf16* __restrict__ wt) {
  int tid = blockIdx.x * 256 + threadIdx.x;      // 0 .. 5*16384-1
  int mat = tid >> 14;
  int idx = tid & 16383;
  int o = idx >> 7, in = idx & 127;
  const float* W = (mat == 0) ? Wq : (mat == 1) ? Wk : (mat == 2) ? Wv
                   : (mat == 3) ? Wg : Wo;
  wt[tid] = (__bf16)W[in * PAIR + o];
}

// ---- 2) fused LayerNorm + QKVG projections ---------------------------------
__global__ __launch_bounds__(256) void proj_kernel(
    const float* __restrict__ z, const float* __restrict__ ln_w,
    const float* __restrict__ ln_b, const __bf16* __restrict__ wt,
    const float* __restrict__ bg, __bf16* __restrict__ qbuf,
    __bf16* __restrict__ kbuf, __bf16* __restrict__ vtbuf,
    __bf16* __restrict__ gbuf) {
  __shared__ __attribute__((aligned(16))) __bf16 znb[128 * PAIR];  // 32 KB
  int pb = blockIdx.x;                 // 1152 blocks of 128 positions
  size_t pos0 = (size_t)pb * 128;
  int ii = pb / 3;                     // 384 = 3 * 128, block within one i
  int j0 = (pb % 3) * 128;
  int tid = threadIdx.x, lane = tid & 31, w = tid >> 5;

  // LayerNorm: 2 threads per row, 64 f32 each, pair-shuffle reduce
  {
    int row = tid >> 1, half = tid & 1;
    const float* zr = z + (pos0 + row) * PAIR + half * 64;
    float s = 0.f, ss = 0.f;
    f32x4 v[16];
#pragma unroll
    for (int u = 0; u < 16; ++u) {
      v[u] = *(const f32x4*)(zr + u * 4);
#pragma unroll
      for (int e = 0; e < 4; ++e) { s += v[u][e]; ss += v[u][e] * v[u][e]; }
    }
    s  += __shfl_xor(s, 1);
    ss += __shfl_xor(ss, 1);
    float mu = s * (1.0f / 128.0f);
    float var = ss * (1.0f / 128.0f) - mu * mu;
    float rstd = rsqrtf(var + 1e-5f);
    __bf16* o = znb + row * PAIR + half * 64;
#pragma unroll
    for (int u = 0; u < 16; ++u)
#pragma unroll
      for (int e = 0; e < 4; ++e) {
        int c = half * 64 + u * 4 + e;
        o[u * 4 + e] = (__bf16)((v[u][e] - mu) * rstd * ln_w[c] + ln_b[c]);
      }
  }
  __syncthreads();

  // GEMM: wave w owns m-tile w (16 rows); 4 mats x 8 n-tiles x 4 k-steps
  bf16x16 af[4];
#pragma unroll
  for (int kk = 0; kk < 4; ++kk)
    af[kk] = frag_ld(znb + (w * 16) * PAIR + kk * 32, PAIR, lane);
  const float qscale = 0.17677669529663687f;  // 32^-0.5
#pragma unroll
  for (int mat = 0; mat < 4; ++mat) {
    const __bf16* wm = wt + mat * 16384;
    for (int nt = 0; nt < 8; ++nt) {
      f32x8 acc = {};
#pragma unroll
      for (int kk = 0; kk < 4; ++kk)
        acc = wmma_bf16(af[kk], frag_ld(wm + (nt * 16) * PAIR + kk * 32, PAIR, lane), acc);
      int nloc = nt * 16 + (lane & 15);
      int hh = nloc >> 5, cc = nloc & 31;
      float bgv = bg[nloc];
#pragma unroll
      for (int r = 0; r < 8; ++r) {
        int rowloc = w * 16 + r + 8 * (lane >> 4);
        int j = j0 + rowloc;
        if (mat == 0)
          qbuf[(((size_t)ii * NHEADS + hh) * NN + j) * HS + cc] = (__bf16)(acc[r] * qscale);
        else if (mat == 1)
          kbuf[(((size_t)ii * NHEADS + hh) * NN + j) * HS + cc] = (__bf16)acc[r];
        else if (mat == 2)
          vtbuf[(((size_t)ii * NHEADS + hh) * HS + cc) * NN + j] = (__bf16)acc[r];
        else {
          float sv = acc[r] + bgv;
          gbuf[(pos0 + rowloc) * PAIR + nloc] = (__bf16)(1.0f / (1.0f + __expf(-sv)));
        }
      }
    }
  }
}

// ---- 3) attention per (i, head, 32-row q-block) ----------------------------
__global__ __launch_bounds__(256) void attn_kernel(
    const __bf16* __restrict__ qbuf, const __bf16* __restrict__ kbuf,
    const __bf16* __restrict__ vtbuf, const __bf16* __restrict__ gbuf,
    const float* __restrict__ mask, __bf16* __restrict__ ogbuf) {
  __shared__ __attribute__((aligned(16))) __bf16 Qs[32 * HS];   // 2 KB
  __shared__ __attribute__((aligned(16))) float  Ss[32 * NN];   // 48 KB
  int qb = blockIdx.x, h = blockIdx.y, i = blockIdx.z;
  int tid = threadIdx.x, lane = tid & 31, w = tid >> 5;
  int q0 = qb * 32;
  const __bf16* Qg = qbuf + (((size_t)i * NHEADS + h) * NN + q0) * HS;
  const __bf16* Kg = kbuf + ((size_t)i * NHEADS + h) * NN * HS;
  const __bf16* Vt = vtbuf + ((size_t)i * NHEADS + h) * HS * NN;

  if (tid < 128) ((f32x4*)Qs)[tid] = ((const f32x4*)Qg)[tid];  // 2 KB copy
  __syncthreads();

  // S = Q K^T + bias : 2 m-tiles x 24 n-tiles, 6 tiles per wave
  {
    int mt = w & 1;
    bf16x16 afrag = frag_ld(Qs + mt * 16 * HS, HS, lane);
    int nt0 = (w >> 1) * 6;
    for (int t = 0; t < 6; ++t) {
      int nt = nt0 + t;
      f32x8 acc = {};
      acc = wmma_bf16(afrag, frag_ld(Kg + nt * 16 * HS, HS, lane), acc);
      int n = nt * 16 + (lane & 15);
      float mb = 1e9f * (mask[(size_t)i * NN + n] - 1.0f);
      float* sp = Ss + (mt * 16 + 8 * (lane >> 4)) * NN + n;
#pragma unroll
      for (int r = 0; r < 8; ++r) sp[r * NN] = acc[r] + mb;
    }
  }
  __syncthreads();

  // softmax over 384 cols: 8 threads per row, shuffle-reduce across 8 lanes
  {
    int row = tid >> 3, part = tid & 7;
    float* rp = Ss + row * NN + part * 48;
    float mx = -3.0e38f;
    for (int c = 0; c < 48; ++c) mx = fmaxf(mx, rp[c]);
    mx = fmaxf(mx, __shfl_xor(mx, 1));
    mx = fmaxf(mx, __shfl_xor(mx, 2));
    mx = fmaxf(mx, __shfl_xor(mx, 4));
    float sum = 0.f;
    for (int c = 0; c < 48; ++c) { float e = __expf(rp[c] - mx); rp[c] = e; sum += e; }
    sum += __shfl_xor(sum, 1);
    sum += __shfl_xor(sum, 2);
    sum += __shfl_xor(sum, 4);
    float inv = 1.0f / sum;
    for (int c = 0; c < 48; ++c) rp[c] *= inv;
  }
  __syncthreads();

  // O = P V  (P f32 in LDS -> bf16 A-frags; B-frags from V^T), then gate
  if (w < 4) {
    int mt = w & 1, ct = w >> 1;
    f32x8 acc = {};
    for (int kk = 0; kk < 12; ++kk) {
      bf16x16 af = frag_ld_f32(Ss + mt * 16 * NN + kk * 32, NN, lane);
      bf16x16 bf = frag_ld(Vt + ct * 16 * NN + kk * 32, NN, lane);
      acc = wmma_bf16(af, bf, acc);
    }
    int c = ct * 16 + (lane & 15);
    int ch = h * HS + c;
#pragma unroll
    for (int r = 0; r < 8; ++r) {
      int qrow = q0 + mt * 16 + r + 8 * (lane >> 4);
      size_t pos = (size_t)i * NN + qrow;
      float g = (float)gbuf[pos * PAIR + ch];
      ogbuf[pos * PAIR + ch] = (__bf16)(acc[r] * g);
    }
  }
}

// ---- 4) output projection: og @ Wo + bo, * mask ----------------------------
__global__ __launch_bounds__(256) void out_kernel(
    const __bf16* __restrict__ ogbuf, const __bf16* __restrict__ wto,
    const float* __restrict__ bo, const float* __restrict__ mask,
    float* __restrict__ out) {
  __shared__ __attribute__((aligned(16))) __bf16 os[128 * PAIR];  // 32 KB
  int pb = blockIdx.x;
  size_t pos0 = (size_t)pb * 128;
  int tid = threadIdx.x, lane = tid & 31, w = tid >> 5;
  const f32x4* src = (const f32x4*)(ogbuf + pos0 * PAIR);
  f32x4* dst = (f32x4*)os;
#pragma unroll
  for (int u = 0; u < 8; ++u) dst[tid + u * 256] = src[tid + u * 256];
  __syncthreads();

  bf16x16 af[4];
#pragma unroll
  for (int kk = 0; kk < 4; ++kk)
    af[kk] = frag_ld(os + (w * 16) * PAIR + kk * 32, PAIR, lane);
  for (int nt = 0; nt < 8; ++nt) {
    f32x8 acc = {};
#pragma unroll
    for (int kk = 0; kk < 4; ++kk)
      acc = wmma_bf16(af[kk], frag_ld(wto + (nt * 16) * PAIR + kk * 32, PAIR, lane), acc);
    int n = nt * 16 + (lane & 15);
    float bov = bo[n];
#pragma unroll
    for (int r = 0; r < 8; ++r) {
      size_t pos = pos0 + w * 16 + r + 8 * (lane >> 4);
      out[pos * PAIR + n] = (acc[r] + bov) * mask[pos];
    }
  }
}

// ---- launcher --------------------------------------------------------------
extern "C" void kernel_launch(void* const* d_in, const int* in_sizes, int n_in,
                              void* d_out, int out_size, void* d_ws, size_t ws_size,
                              hipStream_t stream) {
  (void)in_sizes; (void)n_in; (void)out_size; (void)ws_size;
  const float* z    = (const float*)d_in[0];
  const float* mask = (const float*)d_in[1];
  const float* ln_w = (const float*)d_in[2];
  const float* ln_b = (const float*)d_in[3];
  const float* Wq   = (const float*)d_in[4];
  const float* Wk   = (const float*)d_in[5];
  const float* Wv   = (const float*)d_in[6];
  const float* Wg   = (const float*)d_in[7];
  const float* bg   = (const float*)d_in[8];
  const float* Wo   = (const float*)d_in[9];
  const float* bo   = (const float*)d_in[10];
  float* out = (float*)d_out;

  char* ws = (char*)d_ws;
  const size_t BUF = (size_t)NPOS * PAIR * sizeof(__bf16);  // 37748736
  __bf16* wt    = (__bf16*)ws;                 // 5 * 16384 bf16 = 160 KB
  __bf16* qbuf  = (__bf16*)(ws + 163840);
  __bf16* kbuf  = (__bf16*)(ws + 163840 + BUF);
  __bf16* vtbuf = (__bf16*)(ws + 163840 + 2 * BUF);
  __bf16* gbuf  = (__bf16*)(ws + 163840 + 3 * BUF);
  __bf16* ogbuf = (__bf16*)(ws + 163840 + 4 * BUF);

  prep_kernel<<<320, 256, 0, stream>>>(Wq, Wk, Wv, Wg, Wo, wt);
  proj_kernel<<<NPOS / 128, 256, 0, stream>>>(z, ln_w, ln_b, wt, bg,
                                              qbuf, kbuf, vtbuf, gbuf);
  attn_kernel<<<dim3(NN / 32, NHEADS, NN), 256, 0, stream>>>(qbuf, kbuf, vtbuf,
                                                             gbuf, mask, ogbuf);
  out_kernel<<<NPOS / 128, 256, 0, stream>>>(ogbuf, wt + 4 * 16384, bo, mask, out);
}
